// CausalSelfAttention_58875411694046
// MI455X (gfx1250) — matrix operand order repaired and checked by reading, not compiled, not run on hardware
//
#include <hip/hip_runtime.h>

typedef __attribute__((ext_vector_type(16))) __bf16 v16bf;
typedef __attribute__((ext_vector_type(8)))  __bf16 v8bf;
typedef __attribute__((ext_vector_type(8)))  float  v8f;

#define WMMA_BF16(A, Bm, Cm) \
  __builtin_amdgcn_wmma_f32_16x16x32_bf16(false, (A), false, (Bm), (short)0, (Cm), false, false)

static __device__ __forceinline__ v16bf cat8(v8bf lo, v8bf hi) {
  v16bf r;
#pragma unroll
  for (int i = 0; i < 8; ++i) { r[i] = lo[i]; r[i + 8] = hi[i]; }
  return r;
}

// ---------------------------------------------------------------------------
// Kernel 1: QKV projection.  X[8192,1024] @ Wqkv[1024,3072] + b, scatter to
// Q/K/V bf16 tensors in [B,H,T,64] layout.  Q pre-scaled by 1/sqrt(64).
// Tile: 128(M) x 128(N) x 64(K), 8 waves, wave tile 64x32 (4x2 WMMA frags).
// ---------------------------------------------------------------------------
__global__ __launch_bounds__(256) void qkv_gemm_kernel(
    const float* __restrict__ x, const float* __restrict__ Wqkv,
    const float* __restrict__ bqkv,
    __bf16* __restrict__ Qd, __bf16* __restrict__ Kd, __bf16* __restrict__ Vd) {
  __shared__ __bf16 As[128][72];   // [m][k]
  __shared__ __bf16 Bs[128][72];   // transposed: [n][k]
  const int tid  = threadIdx.x;
  const int lane = tid & 31;
  const int w    = tid >> 5;     // 0..7
  const int wm   = w >> 2;       // 0..1
  const int wn   = w & 3;        // 0..3
  const int h16  = lane >> 4;    // 0/1
  const int l16  = lane & 15;
  const int m0 = blockIdx.y * 128;
  const int n0 = blockIdx.x * 128;

  v8f acc[4][2];
#pragma unroll
  for (int i = 0; i < 4; ++i)
#pragma unroll
    for (int j = 0; j < 2; ++j) acc[i][j] = (v8f)(0.0f);

  for (int k0 = 0; k0 < 1024; k0 += 64) {
    // --- load A tile (fp32 -> bf16), 128x64
#pragma unroll
    for (int p = 0; p < 8; ++p) {
      int idx = tid + p * 256;                 // 0..2047
      int row = idx >> 4;                      // 16 float4 per row
      int c4  = (idx & 15) << 2;
      const float4 f = *(const float4*)(x + (size_t)(m0 + row) * 1024 + k0 + c4);
      __bf16* d = &As[row][c4];
      d[0] = (__bf16)f.x; d[1] = (__bf16)f.y; d[2] = (__bf16)f.z; d[3] = (__bf16)f.w;
    }
    // --- load B tile transposed (fp32 -> bf16), 64x128 -> Bs[n][k]
#pragma unroll
    for (int p = 0; p < 8; ++p) {
      int idx = tid + p * 256;                 // 0..2047
      int kk  = idx >> 5;                      // 32 float4 per row of 128
      int c4  = (idx & 31) << 2;
      const float4 f = *(const float4*)(Wqkv + (size_t)(k0 + kk) * 3072 + n0 + c4);
      Bs[c4 + 0][kk] = (__bf16)f.x; Bs[c4 + 1][kk] = (__bf16)f.y;
      Bs[c4 + 2][kk] = (__bf16)f.z; Bs[c4 + 3][kk] = (__bf16)f.w;
    }
    __syncthreads();
#pragma unroll
    for (int kk = 0; kk < 64; kk += 32) {
      v16bf af[4], bf_[2];
#pragma unroll
      for (int i = 0; i < 4; ++i) {
        const __bf16* pa = &As[wm * 64 + i * 16 + l16][kk + 8 * h16];
        af[i] = cat8(*(const v8bf*)pa, *(const v8bf*)(pa + 16));
      }
#pragma unroll
      for (int j = 0; j < 2; ++j) {
        const __bf16* pb = &Bs[wn * 32 + j * 16 + l16][kk + 8 * h16];
        bf_[j] = cat8(*(const v8bf*)pb, *(const v8bf*)(pb + 16));
      }
#pragma unroll
      for (int i = 0; i < 4; ++i)
#pragma unroll
        for (int j = 0; j < 2; ++j) acc[i][j] = WMMA_BF16(af[i], bf_[j], acc[i][j]);
    }
    __syncthreads();
  }
  // --- epilogue: bias, scale Q, scatter into [B,H,T,64] bf16
#pragma unroll
  for (int i = 0; i < 4; ++i) {
#pragma unroll
    for (int j = 0; j < 2; ++j) {
      int col = n0 + wn * 32 + j * 16 + l16;
      float bias = bqkv[col];
      int which = col >> 10;        // 0:q 1:k 2:v (uniform per fragment)
      int c = col & 1023;
      int hh = c >> 6, dd = c & 63;
      __bf16* dst = (which == 0) ? Qd : ((which == 1) ? Kd : Vd);
      float scale = (which == 0) ? 0.125f : 1.0f;
#pragma unroll
      for (int r = 0; r < 8; ++r) {
        int row = m0 + wm * 64 + i * 16 + r + 8 * h16;   // = b*T + t
        int bb = row >> 11, tt = row & 2047;
        size_t di = (((size_t)(bb * 16 + hh) * 2048) + tt) * 64 + dd;
        dst[di] = (__bf16)((acc[i][j][r] + bias) * scale);
      }
    }
  }
}

// ---------------------------------------------------------------------------
// Kernel 2: flash attention.  One block per (b,h,q-tile of 64).  4 waves,
// each owning 16 q rows.  fp32 online softmax, bf16 WMMA for QK^T and PV.
// Output written bf16 to A2 in [B*T, C] layout for the final projection.
// ---------------------------------------------------------------------------
__global__ __launch_bounds__(128) void attn_kernel(
    const __bf16* __restrict__ Qd, const __bf16* __restrict__ Kd,
    const __bf16* __restrict__ Vd, __bf16* __restrict__ A2) {
  __shared__ __bf16 Kt[64][72];        // [key][d]
  __shared__ __bf16 Vt[64][72];        // transposed: [d][key]
  __shared__ __bf16 Pw[4][16][72];     // per-wave P tile [q][key]
  const int tid  = threadIdx.x;
  const int lane = tid & 31;
  const int w    = tid >> 5;           // 0..3
  const int h16  = lane >> 4;
  const int l16  = lane & 15;
  const int qt0 = blockIdx.x * 64;
  const int bh  = blockIdx.y;
  const int bb  = bh >> 4, hh = bh & 15;
  const size_t headOff = (size_t)(bb * 16 + hh) * 2048 * 64;
  const __bf16* Qh = Qd + headOff;
  const __bf16* Kh = Kd + headOff;
  const __bf16* Vh = Vd + headOff;

  // Q fragments for this wave's 16 rows (d = 0..63 in two K=32 chunks)
  v16bf aq[2];
  {
    const __bf16* pq = Qh + (size_t)(qt0 + w * 16 + l16) * 64;
#pragma unroll
    for (int dk = 0; dk < 2; ++dk) {
      const __bf16* p0 = pq + dk * 32 + 8 * h16;
      aq[dk] = cat8(*(const v8bf*)p0, *(const v8bf*)(p0 + 16));
    }
  }

  float mrow[8], lrow[8];
  v8f accO[4];
#pragma unroll
  for (int r = 0; r < 8; ++r) { mrow[r] = -1.0e30f; lrow[r] = 0.0f; }
#pragma unroll
  for (int fd = 0; fd < 4; ++fd) accO[fd] = (v8f)(0.0f);

  for (int j0 = 0; j0 <= qt0; j0 += 64) {
    __syncthreads();
    // load K tile straight, V tile transposed
#pragma unroll
    for (int p = 0; p < 4; ++p) {
      int idx = tid + p * 128;               // 0..511
      int key = idx >> 3;
      int c8  = (idx & 7) * 8;
      v8bf kv = *(const v8bf*)(Kh + (size_t)(j0 + key) * 64 + c8);
      *(v8bf*)&Kt[key][c8] = kv;
      v8bf vv = *(const v8bf*)(Vh + (size_t)(j0 + key) * 64 + c8);
#pragma unroll
      for (int e = 0; e < 8; ++e) Vt[c8 + e][key] = vv[e];
    }
    __syncthreads();

    // S = Q K^T (Q pre-scaled by 1/sqrt(dk))
    v8f sacc[4];
#pragma unroll
    for (int fn = 0; fn < 4; ++fn) sacc[fn] = (v8f)(0.0f);
#pragma unroll
    for (int kk = 0; kk < 64; kk += 32) {
      v16bf ak = aq[kk >> 5];
#pragma unroll
      for (int fn = 0; fn < 4; ++fn) {
        const __bf16* pb = &Kt[fn * 16 + l16][kk + 8 * h16];
        v16bf bk = cat8(*(const v8bf*)pb, *(const v8bf*)(pb + 16));
        sacc[fn] = WMMA_BF16(ak, bk, sacc[fn]);
      }
    }
    // causal mask on the diagonal tile
    if (j0 == qt0) {
#pragma unroll
      for (int fn = 0; fn < 4; ++fn)
#pragma unroll
        for (int r = 0; r < 8; ++r) {
          int key = j0 + fn * 16 + l16;
          int q   = qt0 + w * 16 + r + 8 * h16;
          if (key > q) sacc[fn][r] = -1.0e30f;
        }
    }
    // online softmax + P to LDS
#pragma unroll
    for (int r = 0; r < 8; ++r) {
      float v = fmaxf(fmaxf(sacc[0][r], sacc[1][r]), fmaxf(sacc[2][r], sacc[3][r]));
      v = fmaxf(v, __shfl_xor(v, 1, 32));
      v = fmaxf(v, __shfl_xor(v, 2, 32));
      v = fmaxf(v, __shfl_xor(v, 4, 32));
      v = fmaxf(v, __shfl_xor(v, 8, 32));
      float newm  = fmaxf(mrow[r], v);
      float alpha = __expf(mrow[r] - newm);
      mrow[r] = newm;
      float ps[4], rs = 0.0f;
#pragma unroll
      for (int fn = 0; fn < 4; ++fn) { ps[fn] = __expf(sacc[fn][r] - newm); rs += ps[fn]; }
      rs += __shfl_xor(rs, 1, 32);
      rs += __shfl_xor(rs, 2, 32);
      rs += __shfl_xor(rs, 4, 32);
      rs += __shfl_xor(rs, 8, 32);
      lrow[r] = lrow[r] * alpha + rs;
#pragma unroll
      for (int fd = 0; fd < 4; ++fd) accO[fd][r] *= alpha;
      int qrow = r + 8 * h16;
#pragma unroll
      for (int fn = 0; fn < 4; ++fn) Pw[w][qrow][fn * 16 + l16] = (__bf16)ps[fn];
    }
    // O += P V  (P re-read from LDS in A-fragment layout; per-wave, in-order LDS)
#pragma unroll
    for (int kk = 0; kk < 64; kk += 32) {
      const __bf16* pp = &Pw[w][l16][kk + 8 * h16];
      v16bf ap = cat8(*(const v8bf*)pp, *(const v8bf*)(pp + 16));
#pragma unroll
      for (int fd = 0; fd < 4; ++fd) {
        const __bf16* pv = &Vt[fd * 16 + l16][kk + 8 * h16];
        v16bf bv = cat8(*(const v8bf*)pv, *(const v8bf*)(pv + 16));
        accO[fd] = WMMA_BF16(ap, bv, accO[fd]);
      }
    }
  }
  // epilogue: normalize, write bf16 to A2 [B*T, C]
#pragma unroll
  for (int r = 0; r < 8; ++r) {
    float inv = 1.0f / lrow[r];
    int q = qt0 + w * 16 + r + 8 * h16;
    size_t mrowg = (size_t)bb * 2048 + q;
#pragma unroll
    for (int fd = 0; fd < 4; ++fd) {
      int col = hh * 64 + fd * 16 + l16;
      A2[mrowg * 1024 + col] = (__bf16)(accO[fd][r] * inv);
    }
  }
}

// ---------------------------------------------------------------------------
// Kernel 3: output projection.  A2[8192,1024](bf16) @ Wout[1024,1024] + b,
// fp32 result to d_out.  Same tiling as kernel 1.
// ---------------------------------------------------------------------------
__global__ __launch_bounds__(256) void out_gemm_kernel(
    const __bf16* __restrict__ A2, const float* __restrict__ Wout,
    const float* __restrict__ bout, float* __restrict__ out) {
  __shared__ __bf16 As[128][72];
  __shared__ __bf16 Bs[128][72];
  const int tid  = threadIdx.x;
  const int lane = tid & 31;
  const int w    = tid >> 5;
  const int wm   = w >> 2;
  const int wn   = w & 3;
  const int h16  = lane >> 4;
  const int l16  = lane & 15;
  const int m0 = blockIdx.y * 128;
  const int n0 = blockIdx.x * 128;

  v8f acc[4][2];
#pragma unroll
  for (int i = 0; i < 4; ++i)
#pragma unroll
    for (int j = 0; j < 2; ++j) acc[i][j] = (v8f)(0.0f);

  for (int k0 = 0; k0 < 1024; k0 += 64) {
    // A tile: already bf16, 128x64 = 1024 16B loads
#pragma unroll
    for (int p = 0; p < 4; ++p) {
      int idx = tid + p * 256;
      int row = idx >> 3;
      int c8  = (idx & 7) * 8;
      *(v8bf*)&As[row][c8] = *(const v8bf*)(A2 + (size_t)(m0 + row) * 1024 + k0 + c8);
    }
    // B tile: fp32 -> bf16 transposed
#pragma unroll
    for (int p = 0; p < 8; ++p) {
      int idx = tid + p * 256;
      int kk  = idx >> 5;
      int c4  = (idx & 31) << 2;
      const float4 f = *(const float4*)(Wout + (size_t)(k0 + kk) * 1024 + n0 + c4);
      Bs[c4 + 0][kk] = (__bf16)f.x; Bs[c4 + 1][kk] = (__bf16)f.y;
      Bs[c4 + 2][kk] = (__bf16)f.z; Bs[c4 + 3][kk] = (__bf16)f.w;
    }
    __syncthreads();
#pragma unroll
    for (int kk = 0; kk < 64; kk += 32) {
      v16bf af[4], bf_[2];
#pragma unroll
      for (int i = 0; i < 4; ++i) {
        const __bf16* pa = &As[wm * 64 + i * 16 + l16][kk + 8 * h16];
        af[i] = cat8(*(const v8bf*)pa, *(const v8bf*)(pa + 16));
      }
#pragma unroll
      for (int j = 0; j < 2; ++j) {
        const __bf16* pb = &Bs[wn * 32 + j * 16 + l16][kk + 8 * h16];
        bf_[j] = cat8(*(const v8bf*)pb, *(const v8bf*)(pb + 16));
      }
#pragma unroll
      for (int i = 0; i < 4; ++i)
#pragma unroll
        for (int j = 0; j < 2; ++j) acc[i][j] = WMMA_BF16(af[i], bf_[j], acc[i][j]);
    }
    __syncthreads();
  }
#pragma unroll
  for (int i = 0; i < 4; ++i) {
#pragma unroll
    for (int j = 0; j < 2; ++j) {
      int col = n0 + wn * 32 + j * 16 + l16;
      float bias = bout[col];
#pragma unroll
      for (int r = 0; r < 8; ++r) {
        int row = m0 + wm * 64 + i * 16 + r + 8 * h16;
        out[(size_t)row * 1024 + col] = acc[i][j][r] + bias;
      }
    }
  }
}

// ---------------------------------------------------------------------------
extern "C" void kernel_launch(void* const* d_in, const int* in_sizes, int n_in,
                              void* d_out, int out_size, void* d_ws, size_t ws_size,
                              hipStream_t stream) {
  (void)in_sizes; (void)n_in; (void)out_size; (void)ws_size;
  const float* x    = (const float*)d_in[0];
  const float* Wqkv = (const float*)d_in[1];
  const float* bqkv = (const float*)d_in[2];
  const float* Wout = (const float*)d_in[3];
  const float* bout = (const float*)d_in[4];
  float* out = (float*)d_out;

  const size_t tensorBytes = (size_t)4 * 16 * 2048 * 64 * sizeof(__bf16); // 16 MiB
  char* ws = (char*)d_ws;
  __bf16* Qd = (__bf16*)(ws);
  __bf16* Kd = (__bf16*)(ws + tensorBytes);
  __bf16* Vd = (__bf16*)(ws + 2 * tensorBytes);
  __bf16* A2 = (__bf16*)(ws + 3 * tensorBytes);

  qkv_gemm_kernel<<<dim3(3072 / 128, 8192 / 128), 256, 0, stream>>>(x, Wqkv, bqkv, Qd, Kd, Vd);
  attn_kernel<<<dim3(2048 / 64, 4 * 16), 128, 0, stream>>>(Qd, Kd, Vd, A2);
  out_gemm_kernel<<<dim3(1024 / 128, 8192 / 128), 256, 0, stream>>>(A2, Wout, bout, out);
}
